// HydraAttention_30562987278515
// MI455X (gfx1250) — matrix operand order repaired and checked by reading, not compiled, not run on hardware
//
#include <hip/hip_runtime.h>
#include <hip/hip_bf16.h>

// ---------------------------------------------------------------------------
// HydraAttention on MI455X (gfx1250). All GEMM-shaped work (QKV proj, KV/QKV
// contractions, output proj, both 3x3 convs as implicit GEMM) runs on
// V_WMMA_F32_16X16X32_BF16 (bf16 in, f32 accumulate), wave32, 8 waves/block,
// 128x128 tile, BK=32, double-buffered LDS staging, pre-converted bf16
// weights, split-K + f32 global atomics for the skinny kv^T contraction.
// Monomorphized via templates to avoid spill-inducing dual-path registers.
// ---------------------------------------------------------------------------

typedef __attribute__((ext_vector_type(16))) __bf16 v16bf;
typedef __attribute__((ext_vector_type(8)))  float  v8f;

#define BM 128
#define BN 128
#define BK 32
#define APAD 40   // ushorts per A-tile row (80B: 16B-aligned rows, bank-skewed)
#define BPAD 40

struct Bits256 { uint4 lo, hi; };

__device__ __forceinline__ unsigned short f2bf(float f) {
  union { float f; unsigned u; } c; c.f = f;
  unsigned u = c.u;
  unsigned r = u + 0x7fffu + ((u >> 16) & 1u);   // RNE
  return (unsigned short)(r >> 16);
}

// A fragment: 16x32 bf16 per ISA 7.12.2 (lane&15=row, lane>>4=K-phase g;
// per lane K = {8g..8g+7, 16+8g..23+8g}) -> two b128 LDS loads.
__device__ __forceinline__ v16bf load_frag_a(const unsigned short* As, int row0, int lane) {
  int r = row0 + (lane & 15);
  int g = lane >> 4;
  const unsigned short* p = As + r * APAD + 8 * g;
  Bits256 t;
  t.lo = *(const uint4*)(p);
  t.hi = *(const uint4*)(p + 16);
  return __builtin_bit_cast(v16bf, t);
}

// B fragment from N-major LDS tile Bt[n][k]: lane&15=col, lane>>4=K-phase kh;
// per lane K = {8kh..8kh+7, 16+8kh..23+8kh} -> two b128 LDS loads.
__device__ __forceinline__ v16bf load_frag_b(const unsigned short* Bt, int col0, int lane) {
  int c = col0 + (lane & 15);
  int kh = lane >> 4;
  const unsigned short* p = Bt + c * BPAD + 8 * kh;
  Bits256 t;
  t.lo = *(const uint4*)(p);
  t.hi = *(const uint4*)(p + 16);
  return __builtin_bit_cast(v16bf, t);
}

// ---------------------------------------------------------------------------
// Batched GEMM: C[b] = A[b](MxK) * B[b](KxN) (+bias[m]) (+rowAdd[b,m])
//               (*colScale[b,n]).
// ABF16:  A is pre-converted bf16 (row-major MxK); else f32.
// BTRANS: B given as (NxK) row-major (stage is a straight copy).
// ATOMIC: split-K accumulation via global_atomic_add_f32.
// ---------------------------------------------------------------------------
template <int ABF16, int BTRANS, int ATOMIC>
__global__ __launch_bounds__(256)
void gemm_wmma_bf16(const void* __restrict__ Ap,
                    const float* __restrict__ B, float* __restrict__ C,
                    int M, int N, int K,
                    long long sA, long long sB, long long sC,
                    int splitK,
                    const float* __restrict__ bias,
                    const float* __restrict__ rowAdd, long long sRowAdd,
                    const float* __restrict__ colScale, long long sColScale)
{
  const int bz = blockIdx.z / splitK;
  const int kc = blockIdx.z % splitK;
  const int Kc = K / splitK;           // K-chunk this block owns
  const int kbase = kc * Kc;

  const float* Af          = (const float*)Ap + bz * sA;          // used if !ABF16
  const unsigned short* Ab = (const unsigned short*)Ap + bz * sA; // used if ABF16
  B += bz * sB; C += bz * sC;

  const int m0 = blockIdx.y * BM;
  const int n0 = blockIdx.x * BN;
  const int tid  = threadIdx.x;
  const int lane = tid & 31;
  const int wid  = tid >> 5;
  const int wm = (wid >> 2) * 64;
  const int wn = (wid & 3) * 32;

  __shared__ __align__(16) unsigned short As[2][BM * APAD];
  __shared__ __align__(16) unsigned short Bs[2][BN * BPAD];

  const v8f vzero = {0.f,0.f,0.f,0.f,0.f,0.f,0.f,0.f};
  v8f acc[4][2];
  #pragma unroll
  for (int i = 0; i < 4; ++i)
    #pragma unroll
    for (int j = 0; j < 2; ++j) acc[i][j] = vzero;

  float4 ra[4];   // f32 A staging regs   (only live when !ABF16)
  uint4  rab[2];  // bf16 A staging regs  (only live when ABF16)
  float4 rb[4];   // B staging regs

  auto loadTile = [&](int k0) {
    if constexpr (ABF16) {
      #pragma unroll
      for (int p = 0; p < 2; ++p) {
        int idx = p * 256 + tid;            // 512 x 16B chunks
        int r = idx >> 2, c8 = (idx & 3) << 3;
        rab[p] = *(const uint4*)(Ab + (long long)(m0 + r) * K + k0 + c8);
      }
    } else {
      #pragma unroll
      for (int p = 0; p < 4; ++p) {
        int idx = p * 256 + tid;            // 1024 x float4
        int r = idx >> 3, c4 = (idx & 7) << 2;
        ra[p] = *(const float4*)(Af + (long long)(m0 + r) * K + k0 + c4);
      }
    }
    if constexpr (!BTRANS) {
      #pragma unroll
      for (int p = 0; p < 4; ++p) {
        int idx = p * 256 + tid;
        int kk = idx >> 5, n4 = (idx & 31) << 2;
        rb[p] = *(const float4*)(B + (long long)(k0 + kk) * N + (n0 + n4));
      }
    } else {
      #pragma unroll
      for (int p = 0; p < 4; ++p) {
        int idx = p * 256 + tid;
        int nn = idx >> 3, k4 = (idx & 7) << 2;
        rb[p] = *(const float4*)(B + (long long)(n0 + nn) * K + k0 + k4);
      }
    }
  };

  auto storeTile = [&](unsigned short* Ad, unsigned short* Bd) {
    if constexpr (ABF16) {
      #pragma unroll
      for (int p = 0; p < 2; ++p) {
        int idx = p * 256 + tid;
        int r = idx >> 2, c8 = (idx & 3) << 3;
        *(uint4*)(Ad + r * APAD + c8) = rab[p];
      }
    } else {
      #pragma unroll
      for (int p = 0; p < 4; ++p) {
        int idx = p * 256 + tid;
        int r = idx >> 3, c4 = (idx & 7) << 2;
        unsigned short* d = Ad + r * APAD + c4;
        d[0] = f2bf(ra[p].x); d[1] = f2bf(ra[p].y); d[2] = f2bf(ra[p].z); d[3] = f2bf(ra[p].w);
      }
    }
    if constexpr (!BTRANS) {
      #pragma unroll
      for (int p = 0; p < 4; ++p) {
        int idx = p * 256 + tid;
        int kk = idx >> 5, n4 = (idx & 31) << 2;
        Bd[(n4 + 0) * BPAD + kk] = f2bf(rb[p].x);
        Bd[(n4 + 1) * BPAD + kk] = f2bf(rb[p].y);
        Bd[(n4 + 2) * BPAD + kk] = f2bf(rb[p].z);
        Bd[(n4 + 3) * BPAD + kk] = f2bf(rb[p].w);
      }
    } else {
      #pragma unroll
      for (int p = 0; p < 4; ++p) {
        int idx = p * 256 + tid;
        int nn = idx >> 3, k4 = (idx & 7) << 2;
        unsigned short* d = Bd + nn * BPAD + k4;
        d[0] = f2bf(rb[p].x); d[1] = f2bf(rb[p].y); d[2] = f2bf(rb[p].z); d[3] = f2bf(rb[p].w);
      }
    }
  };

  const int ksteps = Kc / BK;
  loadTile(kbase);                 // pipeline prologue
  storeTile(As[0], Bs[0]);

  for (int ks = 0; ks < ksteps; ++ks) {
    const int cur = ks & 1;
    if (ks + 1 < ksteps) {
      int kn = kbase + (ks + 1) * BK;
      if constexpr (!BTRANS) __builtin_prefetch(B + (long long)kn * N + n0, 0, 3);
      else                   __builtin_prefetch(B + (long long)n0 * K + kn, 0, 3);
      loadTile(kn);                              // issue next global loads early
    }
    __syncthreads();                             // As/Bs[cur] visible; other buf free

    v16bf af[4], bfr[2];
    #pragma unroll
    for (int i = 0; i < 4; ++i) af[i] = load_frag_a(As[cur], wm + 16 * i, lane);
    #pragma unroll
    for (int j = 0; j < 2; ++j) bfr[j] = load_frag_b(Bs[cur], wn + 16 * j, lane);
    #pragma unroll
    for (int i = 0; i < 4; ++i)
      #pragma unroll
      for (int j = 0; j < 2; ++j)
        acc[i][j] = __builtin_amdgcn_wmma_f32_16x16x32_bf16(
            false, af[i], false, bfr[j], (short)0, acc[i][j], false, false);

    if (ks + 1 < ksteps) storeTile(As[cur ^ 1], Bs[cur ^ 1]);  // fill other buffer
  }

  // epilogue: D layout -> VGPR v = row v + 8*(lane>>4), col = lane&15
  const int halfOff = (lane >> 4) * 8;
  const int cc = lane & 15;
  #pragma unroll
  for (int i = 0; i < 4; ++i) {
    #pragma unroll
    for (int j = 0; j < 2; ++j) {
      int gmb = m0 + wm + 16 * i + halfOff;
      int gn  = n0 + wn + 16 * j + cc;
      float cs = colScale ? colScale[bz * sColScale + gn] : 1.0f;
      #pragma unroll
      for (int v = 0; v < 8; ++v) {
        int gm = gmb + v;
        float val = acc[i][j][v];
        if (rowAdd) val += rowAdd[bz * sRowAdd + gm];
        if (bias)   val += bias[gm];
        val *= cs;
        long long o = (long long)gm * N + gn;
        if constexpr (ATOMIC) atomicAdd(&C[o], val);   // global_atomic_add_f32
        else                  C[o] = val;
      }
    }
  }
}

// ---------------------------------------------------------------------------
// Implicit-GEMM 3x3 conv (SAME), 256->256 over 64x64, bf16 WMMA, double-
// buffered; weights pre-converted+reorganized to bf16 [tap][o][c].
// Optional fused residual: out = out * resid + resid  (h*x + x).
// ---------------------------------------------------------------------------
__global__ __launch_bounds__(256)
void conv3x3_wmma_bf16(const float* __restrict__ in, const unsigned short* __restrict__ wb,
                       const float* __restrict__ bias,
                       const float* __restrict__ resid, float* __restrict__ out)
{
  const int bz = blockIdx.z;
  const long long bst = 256LL * 4096;
  in += bz * bst; out += bz * bst;
  const float* res = resid ? resid + bz * bst : nullptr;

  const int m0 = blockIdx.y * BM;
  const int n0 = blockIdx.x * BN;
  const int tid  = threadIdx.x;
  const int lane = tid & 31;
  const int wid  = tid >> 5;
  const int wm = (wid >> 2) * 64;
  const int wn = (wid & 3) * 32;

  __shared__ __align__(16) unsigned short As[2][BM * APAD];
  __shared__ __align__(16) unsigned short Bs[2][BN * BPAD];

  const v8f vzero = {0.f,0.f,0.f,0.f,0.f,0.f,0.f,0.f};
  v8f acc[4][2];
  #pragma unroll
  for (int i = 0; i < 4; ++i)
    #pragma unroll
    for (int j = 0; j < 2; ++j) acc[i][j] = vzero;

  uint4 rab[2];
  float rbf[16];

  auto loadTile = [&](int s) {                  // s = tap*8 + ctile
    int tap = s >> 3, ct = s & 7;
    int dy = tap / 3 - 1, dx = tap % 3 - 1;
    #pragma unroll
    for (int p = 0; p < 2; ++p) {
      int idx = p * 256 + tid;
      int r = idx >> 2, c8 = (idx & 3) << 3;
      rab[p] = *(const uint4*)(wb + tap * 65536 + (long long)(m0 + r) * 256 + ct * 32 + c8);
    }
    #pragma unroll
    for (int p = 0; p < 16; ++p) {
      int idx = p * 256 + tid;
      int cc = idx >> 7, pp = idx & 127;
      int pix = n0 + pp;
      int y = (pix >> 6) + dy;
      int x = (pix & 63) + dx;
      float v = 0.0f;
      if ((unsigned)y < 64u && (unsigned)x < 64u)
        v = in[(long long)(ct * 32 + cc) * 4096 + y * 64 + x];
      rbf[p] = v;
    }
  };
  auto storeTile = [&](unsigned short* Ad, unsigned short* Bd) {
    #pragma unroll
    for (int p = 0; p < 2; ++p) {
      int idx = p * 256 + tid;
      int r = idx >> 2, c8 = (idx & 3) << 3;
      *(uint4*)(Ad + r * APAD + c8) = rab[p];
    }
    #pragma unroll
    for (int p = 0; p < 16; ++p) {
      int idx = p * 256 + tid;
      int cc = idx >> 7, pp = idx & 127;
      Bd[pp * BPAD + cc] = f2bf(rbf[p]);
    }
  };

  loadTile(0);
  storeTile(As[0], Bs[0]);
  for (int s = 0; s < 72; ++s) {                // 9 taps x 8 c-tiles
    const int cur = s & 1;
    if (s + 1 < 72) loadTile(s + 1);
    __syncthreads();

    v16bf af[4], bfr[2];
    #pragma unroll
    for (int i = 0; i < 4; ++i) af[i] = load_frag_a(As[cur], wm + 16 * i, lane);
    #pragma unroll
    for (int j = 0; j < 2; ++j) bfr[j] = load_frag_b(Bs[cur], wn + 16 * j, lane);
    #pragma unroll
    for (int i = 0; i < 4; ++i)
      #pragma unroll
      for (int j = 0; j < 2; ++j)
        acc[i][j] = __builtin_amdgcn_wmma_f32_16x16x32_bf16(
            false, af[i], false, bfr[j], (short)0, acc[i][j], false, false);

    if (s + 1 < 72) storeTile(As[cur ^ 1], Bs[cur ^ 1]);
  }

  const int halfOff = (lane >> 4) * 8;
  const int cnl = lane & 15;
  #pragma unroll
  for (int i = 0; i < 4; ++i) {
    #pragma unroll
    for (int j = 0; j < 2; ++j) {
      int gmb = m0 + wm + 16 * i + halfOff;
      int gn  = n0 + wn + 16 * j + cnl;
      #pragma unroll
      for (int v = 0; v < 8; ++v) {
        int gm = gmb + v;
        float val = acc[i][j][v] + bias[gm];
        long long o = (long long)gm * 4096 + gn;
        if (res) { float r = res[o]; val = val * r + r; }   // h*x + x
        out[o] = val;
      }
    }
  }
}

// ---------------------------------------------------------------------------
// Small kernels: weight conversion, zero-fill, reductions
// ---------------------------------------------------------------------------
__global__ __launch_bounds__(256)
void cvt_bf16_kernel(const float* __restrict__ in, unsigned short* __restrict__ out, int n)
{
  int i = blockIdx.x * 256 + threadIdx.x;
  if (i < n) out[i] = f2bf(in[i]);
}

// conv weight reorg: out[tap*65536 + o*256 + c] = bf16(w[(o*256+c)*9 + tap])
__global__ __launch_bounds__(256)
void cvt_convw_kernel(const float* __restrict__ in, unsigned short* __restrict__ out)
{
  int i = blockIdx.x * 256 + threadIdx.x;      // 0..589823
  int tap = i >> 16;                           // /65536
  int rem = i & 65535;
  out[i] = f2bf(in[(long long)rem * 9 + tap]);
}

__global__ __launch_bounds__(256)
void zero_kernel(float* __restrict__ p, int n)
{
  int i = blockIdx.x * 256 + threadIdx.x;
  if (i < n) p[i] = 0.0f;
}

// In-place per-token (column) L2 normalization over 256 channels.
__global__ __launch_bounds__(256)
void colnorm_kernel(float* __restrict__ X)
{
  int n = blockIdx.x * 256 + threadIdx.x;
  long long base = (long long)blockIdx.y * 256 * 4096 + n;
  float ss = 0.0f;
  for (int c = 0; c < 256; ++c) { float v = X[base + (long long)c * 4096]; ss += v * v; }
  float rn = rsqrtf(ss);
  for (int c = 0; c < 256; ++c) X[base + (long long)c * 4096] *= rn;
}

// out[b,row] = sum_n X[b,row,n]
__global__ __launch_bounds__(256)
void rowsum_kernel(const float* __restrict__ X, float* __restrict__ out)
{
  __shared__ float red[256];
  int row = blockIdx.x, b = blockIdx.y, tid = threadIdx.x;
  long long base = ((long long)b * 256 + row) * 4096;
  float s = 0.0f;
  for (int n = tid; n < 4096; n += 256) s += X[base + n];
  red[tid] = s; __syncthreads();
  for (int st = 128; st > 0; st >>= 1) {
    if (tid < st) red[tid] += red[tid + st];
    __syncthreads();
  }
  if (tid == 0) out[b * 256 + row] = red[0];
}

// denom[b,n] = 1 / (4096 + sum_k Qn[b,k,n]*ksum[b,k] + eps)
__global__ __launch_bounds__(256)
void denom_kernel(const float* __restrict__ Qn, const float* __restrict__ ksum,
                  float* __restrict__ denom)
{
  __shared__ float ks[256];
  ks[threadIdx.x] = ksum[blockIdx.y * 256 + threadIdx.x];
  __syncthreads();
  int n = blockIdx.x * 256 + threadIdx.x;
  long long base = (long long)blockIdx.y * 256 * 4096 + n;
  float s = 0.0f;
  for (int k = 0; k < 256; ++k) s += Qn[base + (long long)k * 4096] * ks[k];
  denom[blockIdx.y * 4096 + n] = 1.0f / (4096.0f + s + 1e-6f);
}

// ---------------------------------------------------------------------------
extern "C" void kernel_launch(void* const* d_in, const int* in_sizes, int n_in,
                              void* d_out, int out_size, void* d_ws, size_t ws_size,
                              hipStream_t stream) {
  const float* x   = (const float*)d_in[0];
  const float* qw  = (const float*)d_in[1];
  const float* qb  = (const float*)d_in[2];
  const float* kw  = (const float*)d_in[3];
  const float* kb  = (const float*)d_in[4];
  const float* vw  = (const float*)d_in[5];
  const float* vb  = (const float*)d_in[6];
  const float* rw  = (const float*)d_in[7];
  const float* rb  = (const float*)d_in[8];
  const float* c1w = (const float*)d_in[9];
  const float* c1b = (const float*)d_in[10];
  const float* c2w = (const float*)d_in[11];
  const float* c2b = (const float*)d_in[12];
  float* out = (float*)d_out;

  const long long BIG = 8LL * 256 * 4096;     // elements per big buffer
  float* Qb    = (float*)d_ws;                // Q -> Qn                  32 MB
  float* Kb    = Qb + BIG;                    // K -> Kn -> attn          32 MB
  float* Vb    = Kb + BIG;                    // V -> wv                  32 MB
  float* kvT   = Vb + BIG;                    // kv^T [b][cv][ck]          2 MB
  float* ksum  = kvT + 8LL * 256 * 256;
  float* vsum  = ksum + 8 * 256;
  float* denom = vsum + 8 * 256;              // [b][4096]
  unsigned short* qwb  = (unsigned short*)(denom + 8 * 4096);
  unsigned short* kwb  = qwb + 65536;
  unsigned short* vwb  = kwb + 65536;
  unsigned short* rwb  = vwb + 65536;
  unsigned short* c1wb = rwb + 65536;         // [9][256][256]
  unsigned short* c2wb = c1wb + 589824;

  dim3 blk(256);
  dim3 gBig(4096 / BN, 256 / BM, 8);          // (32, 2, 8)
  const long long sX = 256LL * 4096;
  const int SPLITK = 8;

  // --- weight pre-conversion to bf16 (tiny; amortized over 512-block reuse) ---
  cvt_bf16_kernel<<<256, blk, 0, stream>>>(qw, qwb, 65536);
  cvt_bf16_kernel<<<256, blk, 0, stream>>>(kw, kwb, 65536);
  cvt_bf16_kernel<<<256, blk, 0, stream>>>(vw, vwb, 65536);
  cvt_bf16_kernel<<<256, blk, 0, stream>>>(rw, rwb, 65536);
  cvt_convw_kernel<<<2304, blk, 0, stream>>>(c1w, c1wb);
  cvt_convw_kernel<<<2304, blk, 0, stream>>>(c2w, c2wb);
  zero_kernel<<<2048, blk, 0, stream>>>(kvT, 8 * 256 * 256);

  // Q/K/V = W * xf + b   (bf16 A, plain B)
  gemm_wmma_bf16<1,0,0><<<gBig, blk, 0, stream>>>(qwb, x, Qb, 256, 4096, 256, 0, sX, sX, 1, qb, nullptr, 0, nullptr, 0);
  gemm_wmma_bf16<1,0,0><<<gBig, blk, 0, stream>>>(kwb, x, Kb, 256, 4096, 256, 0, sX, sX, 1, kb, nullptr, 0, nullptr, 0);
  gemm_wmma_bf16<1,0,0><<<gBig, blk, 0, stream>>>(vwb, x, Vb, 256, 4096, 256, 0, sX, sX, 1, vb, nullptr, 0, nullptr, 0);

  // per-token L2 normalization (in place)
  colnorm_kernel<<<dim3(16, 8), blk, 0, stream>>>(Qb);
  colnorm_kernel<<<dim3(16, 8), blk, 0, stream>>>(Kb);

  // ksum = Kn.sum(-1), vsum = V.sum(-1)
  rowsum_kernel<<<dim3(256, 8), blk, 0, stream>>>(Kb, ksum);
  rowsum_kernel<<<dim3(256, 8), blk, 0, stream>>>(Vb, vsum);

  // denom = 1 / (n + Qn^T . ksum + eps)
  denom_kernel<<<dim3(16, 8), blk, 0, stream>>>(Qb, ksum, denom);

  // kv^T[c,k] = sum_n V[c,n]*Kn[k,n]  (f32 A, B transposed, split-K x8 -> 256 WGs)
  gemm_wmma_bf16<0,1,1><<<dim3(256 / BN, 256 / BM, 8 * SPLITK), blk, 0, stream>>>(
      Vb, Kb, kvT, 256, 256, 4096, sX, sX, 65536, SPLITK,
      nullptr, nullptr, 0, nullptr, 0);

  // wv[c,n] = (kv^T . Qn + vsum[c]) * denom[n]  -> Vb (V dead)
  gemm_wmma_bf16<0,0,0><<<gBig, blk, 0, stream>>>(kvT, Qb, Vb, 256, 4096, 256, 65536, sX, sX, 1,
                                                  nullptr, vsum, 256, denom, 4096);

  // attn = rw * wv + rb  -> Kb (K dead)
  gemm_wmma_bf16<1,0,0><<<gBig, blk, 0, stream>>>(rwb, Vb, Kb, 256, 4096, 256, 0, sX, sX, 1,
                                                  rb, nullptr, 0, nullptr, 0);

  // h1 = conv3x3(attn) + c1b  -> Qb (Q dead)
  conv3x3_wmma_bf16<<<gBig, blk, 0, stream>>>(Kb, c1wb, c1b, nullptr, Qb);

  // out = (conv3x3(h1) + c2b) * x + x
  conv3x3_wmma_bf16<<<gBig, blk, 0, stream>>>(Qb, c2wb, c2b, x, out);
}
